// ImportanceRenderer_79834852098352
// MI455X (gfx1250) — matrix operand order repaired and checked by reading, not compiled, not run on hardware
//
#include <hip/hip_runtime.h>
#include <hip/hip_bf16.h>

typedef __attribute__((ext_vector_type(16))) _Float16 v16h;
typedef __attribute__((ext_vector_type(8)))  float    v8f;

namespace {
constexpr int   kB = 2, kR = 4096, kNC = 48, kNI = 48;
constexpr int   kC = 32, kH = 256, kHid = 64;
constexpr int   kNRay = kB * kR;        // 8192
constexpr int   kNPts = kNRay * kNC;    // 393216
constexpr float kRayStart = 0.1f, kRayEnd = 1.0f;
constexpr float kDelta = (kRayEnd - kRayStart) / (kNC - 1);

// Branchless fast softplus: max(x,0) + log(1 + exp(-|x|)).
// Uses hardware v_exp_f32 / v_log_f32 transcendentals.
__device__ __forceinline__ float softplusf(float x) {
  return fmaxf(x, 0.f) + __logf(1.f + __expf(-fabsf(x)));
}
__device__ __forceinline__ float sigmoidf_(float x) {
  return __builtin_amdgcn_rcpf(1.f + __expf(-x));
}
} // namespace

// ---------------------------------------------------------------------------
// Stage A/C: tri-plane gather + 64->64->4 MLP via V_WMMA_F32_16X16X32_F16.
// One wave == one 16-point tile. lane = (point m = lane&15, set hi = lane>>4).
// ---------------------------------------------------------------------------
__global__ __launch_bounds__(128) void featmlp_kernel(
    const float* __restrict__ ptex, const float* __restrict__ pshp,
    const float* __restrict__ rayo, const float* __restrict__ rayd,
    const float* __restrict__ W1, const float* __restrict__ b1,
    const float* __restrict__ W2, const float* __restrict__ b2,
    const float* __restrict__ noise, const float* __restrict__ dfine,
    float* __restrict__ out_sig, float* __restrict__ out_rgb) {
  __shared__ _Float16 w1h[kHid * kHid];  // [K][N]
  __shared__ _Float16 w2h[kHid * 16];    // [K][N], N<4 valid, rest zero
  __shared__ _Float16 xt[4][16 * 64];    // per-wave tile [M][K]; reused for H

  const int tid = threadIdx.x;
  for (int i = tid; i < kHid * kHid; i += 128) w1h[i] = (_Float16)W1[i];
  for (int i = tid; i < kHid * 16; i += 128) {
    int k = i >> 4, n = i & 15;
    w2h[i] = (n < 4) ? (_Float16)W2[k * 4 + n] : (_Float16)0.f;
  }

  const int wave = tid >> 5, lane = tid & 31;
  const int m = lane & 15, hi = lane >> 4;
  const int ptBase = (blockIdx.x * 4 + wave) * 16;

  // ---- tri-plane features for point (ptBase+m), feature set hi ----
  const int pt  = ptBase + m;
  const int b   = pt / (kR * kNC);
  const int rem = pt - b * kR * kNC;
  const int r   = rem / kNC;
  const int si  = rem - r * kNC;
  const int ray = b * kR + r;

  float d = dfine ? dfine[pt] : (kRayStart + kDelta * si + noise[pt] * kDelta);
  float rdx = rayd[ray * 3], rdy = rayd[ray * 3 + 1], rdz = rayd[ray * 3 + 2];
  float inv = rsqrtf(rdx * rdx + rdy * rdy + rdz * rdz);
  float cx = rayo[ray * 3]     + d * rdx * inv;
  float cy = rayo[ray * 3 + 1] + d * rdy * inv;
  float cz = rayo[ray * 3 + 2] + d * rdz * inv;

  const float* plane = hi ? pshp : ptex;
  float gx[3] = {cx, cx, cz};
  float gy[3] = {cy, cz, cx};
  float wgt[3][4];
  int   off[3][4];
#pragma unroll
  for (int p = 0; p < 3; ++p) {
    float x = (gx[p] + 1.f) * (kH * 0.5f) - 0.5f;
    float y = (gy[p] + 1.f) * (kH * 0.5f) - 0.5f;
    float x0f = floorf(x), y0f = floorf(y);
    int x0 = (int)x0f, y0 = (int)y0f;
    float fx = x - x0f, fy = y - y0f;
#pragma unroll
    for (int t = 0; t < 4; ++t) {
      int ddx = t & 1, ddy = t >> 1;
      int xi = x0 + ddx, yi = y0 + ddy;
      float w = (ddx ? fx : 1.f - fx) * (ddy ? fy : 1.f - fy);
      bool valid = (xi >= 0) && (xi < kH) && (yi >= 0) && (yi < kH);
      int xc = min(max(xi, 0), kH - 1), yc = min(max(yi, 0), kH - 1);
      wgt[p][t] = valid ? w : 0.f;
      off[p][t] = yc * kH + xc;
    }
  }
  const long chStride = (long)kH * kH;
  const float* pb = plane + (long)b * 3 * kC * chStride;
  for (int c = 0; c < kC; ++c) {
    float acc = 0.f;
#pragma unroll
    for (int p = 0; p < 3; ++p) {
      const float* fp = pb + ((long)p * kC + c) * chStride;
      acc += wgt[p][0] * fp[off[p][0]] + wgt[p][1] * fp[off[p][1]] +
             wgt[p][2] * fp[off[p][2]] + wgt[p][3] * fp[off[p][3]];
    }
    xt[wave][m * 64 + hi * 32 + c] = (_Float16)(acc * (1.f / 3.f));
  }
  __syncthreads();

  // ---- layer 1: X(16x64) @ W1(64x64) + b1, softplus ----
  v16h a0, a1;  // A-layout fragments, K chunk 0 and 1
#pragma unroll
  for (int v = 0; v < 8; ++v) {
    int ka = hi * 16 + 2 * v;
    a0[2 * v]     = xt[wave][m * 64 + ka];
    a0[2 * v + 1] = xt[wave][m * 64 + ka + 1];
    a1[2 * v]     = xt[wave][m * 64 + 32 + ka];
    a1[2 * v + 1] = xt[wave][m * 64 + 32 + ka + 1];
  }
  float hres[4][8];
#pragma unroll
  for (int nt = 0; nt < 4; ++nt) {
    int col = nt * 16 + m;
    float bias = b1[col];
    v8f acc;
#pragma unroll
    for (int j = 0; j < 8; ++j) acc[j] = bias;
    v16h bf0, bf1;  // B-layout fragments
#pragma unroll
    for (int v = 0; v < 8; ++v) {
      int kb = hi * 16 + 2 * v;
      bf0[2 * v]     = w1h[kb * kHid + col];
      bf0[2 * v + 1] = w1h[(kb + 1) * kHid + col];
      bf1[2 * v]     = w1h[(32 + kb) * kHid + col];
      bf1[2 * v + 1] = w1h[(33 + kb) * kHid + col];
    }
    acc = __builtin_amdgcn_wmma_f32_16x16x32_f16(false, a0, false, bf0,
                                                 (short)0, acc, false, false);
    acc = __builtin_amdgcn_wmma_f32_16x16x32_f16(false, a1, false, bf1,
                                                 (short)0, acc, false, false);
#pragma unroll
    for (int j = 0; j < 8; ++j) hres[nt][j] = softplusf(acc[j]);
  }
  __syncthreads();
  // D-layout (M = j + 8*hi, N = m) -> LDS [M][K] for re-fragmenting as A
#pragma unroll
  for (int nt = 0; nt < 4; ++nt)
#pragma unroll
    for (int j = 0; j < 8; ++j)
      xt[wave][(j + 8 * hi) * 64 + nt * 16 + m] = (_Float16)hres[nt][j];
  __syncthreads();

  // ---- layer 2: H(16x64) @ W2(64x4-pad-16) + b2 ----
  v16h h0, h1;
#pragma unroll
  for (int v = 0; v < 8; ++v) {
    int ka = hi * 16 + 2 * v;
    h0[2 * v]     = xt[wave][m * 64 + ka];
    h0[2 * v + 1] = xt[wave][m * 64 + ka + 1];
    h1[2 * v]     = xt[wave][m * 64 + 32 + ka];
    h1[2 * v + 1] = xt[wave][m * 64 + 32 + ka + 1];
  }
  v16h c0, c1;
#pragma unroll
  for (int v = 0; v < 8; ++v) {
    int kb = hi * 16 + 2 * v;
    c0[2 * v]     = w2h[kb * 16 + m];
    c0[2 * v + 1] = w2h[(kb + 1) * 16 + m];
    c1[2 * v]     = w2h[(32 + kb) * 16 + m];
    c1[2 * v + 1] = w2h[(33 + kb) * 16 + m];
  }
  float bias2 = (m < 4) ? b2[m] : 0.f;
  v8f o;
#pragma unroll
  for (int j = 0; j < 8; ++j) o[j] = bias2;
  o = __builtin_amdgcn_wmma_f32_16x16x32_f16(false, h0, false, c0, (short)0, o,
                                             false, false);
  o = __builtin_amdgcn_wmma_f32_16x16x32_f16(false, h1, false, c1, (short)0, o,
                                             false, false);

  if (m < 4) {
#pragma unroll
    for (int j = 0; j < 8; ++j) {
      int opt = ptBase + j + 8 * hi;
      float val = o[j];
      if (m == 0)
        out_sig[opt] = val;  // raw sigma; softplus applied in march
      else
        out_rgb[opt * 3 + (m - 1)] = 1.002f * sigmoidf_(val) - 0.001f;
    }
  }
}

// ---------------------------------------------------------------------------
// Stage B: coarse ray-march weights + inverse-CDF importance sampling.
// ---------------------------------------------------------------------------
__global__ __launch_bounds__(128) void importance_kernel(
    const float* __restrict__ noise, const float* __restrict__ sig_c,
    const float* __restrict__ imp_u, float* __restrict__ dfine) {
  int ray = blockIdx.x * blockDim.x + threadIdx.x;
  if (ray >= kNRay) return;
  float z[kNC], s[kNC];
  for (int i = 0; i < kNC; ++i) {
    z[i] = kRayStart + kDelta * i + noise[ray * kNC + i] * kDelta;
    s[i] = sig_c[ray * kNC + i];
  }
  float w[kNC - 1];
  float T = 1.f;
  for (int i = 0; i < kNC - 1; ++i) {
    float dd = z[i + 1] - z[i];
    float dens = softplusf(0.5f * (s[i] + s[i + 1]) - 1.f);
    float alpha = 1.f - __expf(-dens * dd);
    w[i] = alpha * T;
    T *= (1.f - alpha + 1e-10f);
  }
  float zmid[kNC - 1];
  for (int i = 0; i < kNC - 1; ++i) zmid[i] = 0.5f * (z[i] + z[i + 1]);
  // wp=[0,w,0]; wmax[i]=max(wp[i],wp[i+1]); wsm=0.5*(wmax[i]+wmax[i+1])+0.01;
  // pdf weights = wsm[1..45]
  float pw[45], cdf[46];
  float tot = 0.f;
  for (int i = 1; i <= 45; ++i) {
    float wp0 = w[i - 1], wp1 = w[i], wp2 = w[i + 1];
    float v = 0.5f * (fmaxf(wp0, wp1) + fmaxf(wp1, wp2)) + 0.01f;
    pw[i - 1] = v;
    tot += v;
  }
  float invTot = 1.f / tot;
  cdf[0] = 0.f;
  for (int i = 1; i <= 45; ++i) cdf[i] = cdf[i - 1] + pw[i - 1] * invTot;
  for (int k = 0; k < kNI; ++k) {
    float u = imp_u[ray * kNI + k];
    int lo = 0, hib = 46;  // searchsorted right over cdf[0..45]
    while (lo < hib) {
      int mid = (lo + hib) >> 1;
      if (cdf[mid] > u) hib = mid; else lo = mid + 1;
    }
    int below = max(lo - 1, 0);
    int above = min(lo, 45);
    float cc0 = cdf[below], cc1 = cdf[above];
    float den = cc1 - cc0;
    if (den < 1e-5f) den = 1.f;
    float t = (u - cc0) / den;
    dfine[ray * kNI + k] = zmid[below] + t * (zmid[above] - zmid[below]);
  }
}

// ---------------------------------------------------------------------------
// Stage D: sort 96 samples per ray, final ray-march, write outputs.
// ---------------------------------------------------------------------------
__device__ __forceinline__ void load_sample(int ray, int j,
                                            const float* __restrict__ sig_c,
                                            const float* __restrict__ rgb_c,
                                            const float* __restrict__ sig_f,
                                            const float* __restrict__ rgb_f,
                                            float& s, float c[3]) {
  if (j < kNC) {
    int p = ray * kNC + j;
    s = sig_c[p]; c[0] = rgb_c[p * 3]; c[1] = rgb_c[p * 3 + 1]; c[2] = rgb_c[p * 3 + 2];
  } else {
    int p = ray * kNI + (j - kNC);
    s = sig_f[p]; c[0] = rgb_f[p * 3]; c[1] = rgb_f[p * 3 + 1]; c[2] = rgb_f[p * 3 + 2];
  }
}

__global__ __launch_bounds__(128) void composite_kernel(
    const float* __restrict__ noise, const float* __restrict__ dfine,
    const float* __restrict__ sig_c, const float* __restrict__ rgb_c,
    const float* __restrict__ sig_f, const float* __restrict__ rgb_f,
    float* __restrict__ out) {
  int ray = blockIdx.x * blockDim.x + threadIdx.x;
  if (ray >= kNRay) return;
  float dall[kNC + kNI];
  int   idx[kNC + kNI];
  for (int i = 0; i < kNC; ++i) {
    dall[i] = kRayStart + kDelta * i + noise[ray * kNC + i] * kDelta;
    idx[i] = i;
  }
  for (int i = 0; i < kNI; ++i) {
    dall[kNC + i] = dfine[ray * kNI + i];
    idx[kNC + i] = kNC + i;
  }
  for (int i = 1; i < kNC + kNI; ++i) {  // stable insertion sort
    float dv = dall[i];
    int iv = idx[i];
    int j = i - 1;
    while (j >= 0 && dall[j] > dv) {
      dall[j + 1] = dall[j]; idx[j + 1] = idx[j]; --j;
    }
    dall[j + 1] = dv; idx[j + 1] = iv;
  }
  float sPrev, cPrev[3];
  load_sample(ray, idx[0], sig_c, rgb_c, sig_f, rgb_f, sPrev, cPrev);
  float T = 1.f, racc[3] = {0.f, 0.f, 0.f}, dacc = 0.f, wacc = 0.f;
  for (int i = 0; i < kNC + kNI - 1; ++i) {
    float sNext, cNext[3];
    load_sample(ray, idx[i + 1], sig_c, rgb_c, sig_f, rgb_f, sNext, cNext);
    float dd = dall[i + 1] - dall[i];
    float zm = 0.5f * (dall[i] + dall[i + 1]);
    float dens = softplusf(0.5f * (sPrev + sNext) - 1.f);
    float alpha = 1.f - __expf(-dens * dd);
    float wgt = alpha * T;
    T *= (1.f - alpha + 1e-10f);
    racc[0] += wgt * 0.5f * (cPrev[0] + cNext[0]);
    racc[1] += wgt * 0.5f * (cPrev[1] + cNext[1]);
    racc[2] += wgt * 0.5f * (cPrev[2] + cNext[2]);
    dacc += wgt * zm;
    wacc += wgt;
    sPrev = sNext;
    cPrev[0] = cNext[0]; cPrev[1] = cNext[1]; cPrev[2] = cNext[2];
  }
  float depth = dacc / wacc;
  if (!isfinite(depth)) depth = 0.f;
  depth = fminf(fmaxf(depth, dall[0]), dall[kNC + kNI - 1]);
  out[ray * 3 + 0] = racc[0] * 2.f - 1.f;
  out[ray * 3 + 1] = racc[1] * 2.f - 1.f;
  out[ray * 3 + 2] = racc[2] * 2.f - 1.f;
  out[kNRay * 3 + ray] = depth;
  out[kNRay * 4 + ray] = wacc;
}

// ---------------------------------------------------------------------------
extern "C" void kernel_launch(void* const* d_in, const int* in_sizes, int n_in,
                              void* d_out, int out_size, void* d_ws,
                              size_t ws_size, hipStream_t stream) {
  (void)in_sizes; (void)n_in; (void)out_size; (void)ws_size;
  const float* ptex  = (const float*)d_in[0];
  const float* pshp  = (const float*)d_in[1];
  const float* rayo  = (const float*)d_in[2];
  const float* rayd  = (const float*)d_in[3];
  const float* W1    = (const float*)d_in[4];
  const float* b1    = (const float*)d_in[5];
  const float* W2    = (const float*)d_in[6];
  const float* b2    = (const float*)d_in[7];
  const float* noise = (const float*)d_in[8];
  const float* imp_u = (const float*)d_in[9];
  float* out = (float*)d_out;

  float* ws    = (float*)d_ws;
  float* sig_c = ws;                 // kNPts
  float* rgb_c = ws + (size_t)kNPts; // 3*kNPts
  float* sig_f = ws + (size_t)4 * kNPts;
  float* rgb_f = ws + (size_t)5 * kNPts;
  float* dfin  = ws + (size_t)8 * kNPts;

  const int mlpBlocks = kNPts / (4 * 16);  // 4 waves x 16 pts per block
  const int rayBlocks = (kNRay + 127) / 128;

  featmlp_kernel<<<mlpBlocks, 128, 0, stream>>>(ptex, pshp, rayo, rayd, W1, b1,
                                                W2, b2, noise, nullptr, sig_c,
                                                rgb_c);
  importance_kernel<<<rayBlocks, 128, 0, stream>>>(noise, sig_c, imp_u, dfin);
  featmlp_kernel<<<mlpBlocks, 128, 0, stream>>>(ptex, pshp, rayo, rayd, W1, b1,
                                                W2, b2, noise, dfin, sig_f,
                                                rgb_f);
  composite_kernel<<<rayBlocks, 128, 0, stream>>>(noise, dfin, sig_c, rgb_c,
                                                  sig_f, rgb_f, out);
}
